// MatchLSTM_22900765622330
// MI455X (gfx1250) — compile-verified, matrix operand order checked by reading, not run on hardware
//
#include <hip/hip_runtime.h>

// ---------------------------------------------------------------------------
// Match-LSTM QA forward for MI455X (gfx1250, wave32, WMMA).
// Sizes: V=50000 E=300 H=150 B=32 LP=400 LQ=50
// Padded: E,2H: 300->320 ; H: 150->160 ; 4H: 600->608
// All matmuls on v_wmma_f32_16x16x32_f16 (f16 in, f32 accumulate).
// Weights pre-packed into WMMA-fragment order so every fragment load is
// 2x b128 per lane (global or LDS) instead of 16 strided u16 loads.
// ---------------------------------------------------------------------------

#define EE   300
#define HH   150
#define BB   32
#define LLP  400
#define LLQ  50

#define EP   320   // E / 2H padded (K dim, multiple of 32)
#define HPD  160   // H padded
#define G4   608   // 4H padded (N dim, multiple of 16)
#define H2P  320   // 2H padded
#define G4T  (G4 / 16)    // 38 N-tiles of the gate matrix
#define HPT  (HPD / 16)   // 10 N-tiles of an H-wide matrix
#define KT_H  (HPD / 32)  // 5 K-tiles over H
#define KT_2H (H2P / 32)  // 10 K-tiles over 2H / E

typedef __attribute__((ext_vector_type(16))) _Float16 v16h;
typedef __attribute__((ext_vector_type(8)))  float    v8f;

__device__ __forceinline__ v8f wmma_f16(v16h a, v16h b, v8f c) {
  return __builtin_amdgcn_wmma_f32_16x16x32_f16(
      /*neg_a=*/false, a, /*neg_b=*/false, b,
      /*c_mod=*/(short)0, c, /*reuse_a=*/false, /*reuse_b=*/false);
}

// Row-major f16 A fragment (contiguous per lane already: 2x b128 loads).
__device__ __forceinline__ v16h load_a16(const _Float16* A, int lda,
                                         int row0, int k0, int lane) {
  const _Float16* p = A + (size_t)(row0 + (lane & 15)) * lda
                        + k0 + ((lane >> 4) << 3);
  v16h a;
#pragma unroll
  for (int i = 0; i < 8; ++i) { a[i] = p[i]; a[i + 8] = p[i + 16]; }
  return a;
}

// Fragment-packed load: tile t0 (m-tile for A / n-tile for B), K-tile kt.
// Each lane reads 16 contiguous halves (32 B) -> two b128 loads.
__device__ __forceinline__ v16h load_frag_pk(const _Float16* P, int ktiles,
                                             int t0, int kt, int lane) {
  const _Float16* p = P + (((size_t)(t0 * ktiles + kt) << 9) + (lane << 4));
  v16h v;
#pragma unroll
  for (int i = 0; i < 16; ++i) v[i] = p[i];
  return v;
}

// Packed-layout flat index for element (row b in [0,32), k-position j).
// lane = (b&15) + 16*sel, sel = (j>>3)&1 ; half i = (j&7) + 8*((j&31)>>4).
__device__ __forceinline__ int apack_idx(int b, int j, int ktiles) {
  int mt = b >> 4, kt = j >> 5, kk = j & 31;
  int lane = (b & 15) + (((kk >> 3) & 1) << 4);
  int i = (kk & 7) + (((kk >> 4) & 1) << 3);
  return ((mt * ktiles + kt) << 9) + (lane << 4) + i;
}

__device__ __forceinline__ float sigmf(float x) {
  return 1.f / (1.f + __expf(-x));
}

// ---------------------------------------------------------------------------
// Fused convert + transpose + fragment-pack of a weight matrix.
// W [n_rows x k_cols] f32  ->  packed f16, tile = nt*ktiles + kt,
// element (lane,i) = W^T[kt*32 + kmap(lane,i)][nt*16 + (lane&15)], zero pad.
// ---------------------------------------------------------------------------
__global__ void k_cvt_pack(const float* __restrict__ W, _Float16* __restrict__ Bpk,
                           int n_rows, int k_cols, int npad, int kpad) {
  int idx = blockIdx.x * blockDim.x + threadIdx.x;
  if (idx >= npad * kpad) return;
  int i    = idx & 15;
  int lane = (idx >> 4) & 31;
  int tile = idx >> 9;
  int ktiles = kpad >> 5;
  int nt = tile / ktiles, kt = tile - nt * ktiles;
  int n = (nt << 4) + (lane & 15);
  int k = (kt << 5) + ((lane >> 4) << 3) + (i & 7) + ((i >> 3) << 4);
  float v = (n < n_rows && k < k_cols) ? W[(size_t)n * k_cols + k] : 0.f;
  Bpk[idx] = (_Float16)v;
}

// Embedding gather -> padded f16 rows [ntok x EP]
__global__ void k_gather(const int* __restrict__ ids, const float* __restrict__ emb,
                         _Float16* __restrict__ out, int ntok) {
  int idx = blockIdx.x * blockDim.x + threadIdx.x;
  if (idx >= ntok * EP) return;
  int t = idx / EP, e = idx - t * EP;
  float v = 0.f;
  if (e < EE) v = emb[(size_t)ids[t] * EE + e];
  out[idx] = (_Float16)v;
}

__global__ void k_zero16(_Float16* __restrict__ p, int n) {
  int i = blockIdx.x * blockDim.x + threadIdx.x;
  if (i < n) p[i] = (_Float16)0.f;
}

// ---------------------------------------------------------------------------
// Generic WMMA GEMM: C[M x Npad] f32 = A[M x Kpad] f16 (row-major)
//                                      * Bpk (fragment-packed W^T) + bias.
// One wave = one 16x16 C tile. blockDim = 256 (8 waves).
// ---------------------------------------------------------------------------
__global__ void k_gemm(const _Float16* __restrict__ A, const _Float16* __restrict__ Bpk,
                       float* __restrict__ C, int M, int Npad, int Kpad,
                       const float* __restrict__ bias, int nValid) {
  const int lane = threadIdx.x & 31;
  const int wave = threadIdx.x >> 5;
  const int tile = blockIdx.x * (blockDim.x >> 5) + wave;
  const int nTiles = Npad >> 4;
  const int kTiles = Kpad >> 5;
  if (tile >= (M >> 4) * nTiles) return;
  const int mt = tile / nTiles, nt = tile - mt * nTiles;
  const int row0 = mt << 4;
  v8f acc = {};
  for (int kt = 0; kt < kTiles; ++kt)
    acc = wmma_f16(load_a16(A, Kpad, row0, kt << 5, lane),
                   load_frag_pk(Bpk, kTiles, nt, kt, lane), acc);
  const int col = (nt << 4) + (lane & 15);
  const int rb  = row0 + ((lane >> 4) << 3);
  const float bv = (bias != nullptr && col < nValid) ? bias[col] : 0.f;
#pragma unroll
  for (int r = 0; r < 8; ++r)
    C[(size_t)(rb + r) * Npad + col] = acc[r] + bv;
}

// ---------------------------------------------------------------------------
// Pre-encoding LSTM scan. grid = 2 (0: passage, 1: question), 1024 threads
// (32 wave32s on one WGP). h lives in LDS in fragment-packed f16 layout;
// c lives in registers (5 elements/thread).
// ---------------------------------------------------------------------------
__global__ void __launch_bounds__(1024)
k_prelstm(const float* __restrict__ XWp, const float* __restrict__ XWq,
          const _Float16* __restrict__ Bhh, const float* __restrict__ bias,
          const int* __restrict__ plens, const int* __restrict__ qlens,
          _Float16* __restrict__ Hp16, _Float16* __restrict__ Hq16,
          float* __restrict__ Hq32) {
  extern __shared__ char smem[];
  _Float16* h16   = (_Float16*)smem;                 // 32*HPD f16, packed
  float*    gates = (float*)(smem + BB * HPD * 2);   // 32*G4 f32

  const bool isQ = (blockIdx.x == 1);
  const float* XW   = isQ ? XWq : XWp;
  const int*   lens = isQ ? qlens : plens;
  const int    L    = isQ ? LLQ : LLP;
  _Float16*    H16  = isQ ? Hq16 : Hp16;
  float*       H32  = isQ ? Hq32 : (float*)0;

  const int tid = threadIdx.x, lane = tid & 31, wave = tid >> 5;
  for (int i = tid; i < BB * HPD; i += 1024) h16[i] = (_Float16)0.f;
  float creg[5];
#pragma unroll
  for (int k = 0; k < 5; ++k) creg[k] = 0.f;
  __syncthreads();

  for (int t = 0; t < L; ++t) {
    const float* xw = XW + (size_t)t * BB * G4;
    // gates = XW[t] + h @ Whh^T + b : 2x38 = 76 tiles over 32 waves
    for (int tile = wave; tile < 2 * G4T; tile += 32) {
      int mt = tile / G4T, nt = tile - mt * G4T;
      v8f acc = {};
#pragma unroll
      for (int kt = 0; kt < KT_H; ++kt)
        acc = wmma_f16(load_frag_pk(h16, KT_H, mt, kt, lane),
                       load_frag_pk(Bhh, KT_H, nt, kt, lane), acc);
      int col = (nt << 4) + (lane & 15);
      int rb  = (mt << 4) + ((lane >> 4) << 3);
      float bv = (col < 4 * HH) ? bias[col] : 0.f;
#pragma unroll
      for (int r = 0; r < 8; ++r)
        gates[(rb + r) * G4 + col] = acc[r] + xw[(size_t)(rb + r) * G4 + col] + bv;
    }
    __syncthreads();
    // LSTM pointwise (torch gate order i,f,g,o)
#pragma unroll
    for (int k = 0; k < 5; ++k) {
      int p = tid + (k << 10);
      if (p < BB * HH) {
        int j = p >> 5, b = p & 31;
        float gi = gates[b * G4 + j];
        float gf = gates[b * G4 + HH + j];
        float gg = gates[b * G4 + 2 * HH + j];
        float go = gates[b * G4 + 3 * HH + j];
        float c2 = sigmf(gf) * creg[k] + sigmf(gi) * tanhf(gg);
        float hv = sigmf(go) * tanhf(c2);
        creg[k] = c2;
        float m  = (t < lens[b]) ? 1.f : 0.f;
        float hm = hv * m;
        size_t orow = ((size_t)t * BB + b) * HPD + j;
        H16[orow] = (_Float16)hm;                   // masked output
        if (H32) H32[orow] = hm;
        h16[apack_idx(b, j, KT_H)] = (_Float16)hv;  // carry unmasked
      }
    }
    __syncthreads();
  }
}

// ---------------------------------------------------------------------------
// Match-LSTM scan. grid = 2 (dir 0: fwd, 1: bwd), 1024 threads.
// ---------------------------------------------------------------------------
__global__ void __launch_bounds__(1024)
k_match(const float* __restrict__ aq, const float* __restrict__ HpWp,
        const float* __restrict__ Hq32, const _Float16* __restrict__ Hp16,
        const _Float16* __restrict__ Bwr,
        const _Float16* __restrict__ Bfih, const _Float16* __restrict__ Bfhh,
        const float* __restrict__ bf,
        const _Float16* __restrict__ Bbih, const _Float16* __restrict__ Bbhh,
        const float* __restrict__ bbk,
        const float* __restrict__ wa, const float* __restrict__ ba,
        const int* __restrict__ plens,
        _Float16* __restrict__ Hr16) {
  extern __shared__ char smem[];
  float*    hWr   = (float*)smem;                                       // 32*HPD
  float*    gates = (float*)(smem + BB * HPD * 4);                      // 32*G4
  float*    sco   = (float*)(smem + BB * HPD * 4 + BB * G4 * 4);        // LQ*32
  _Float16* h16   = (_Float16*)(smem + BB * HPD * 4 + BB * G4 * 4 + LLQ * BB * 4);
  _Float16* z16   = (_Float16*)((char*)h16 + BB * HPD * 2);             // 32*H2P

  const int dir = blockIdx.x;
  const _Float16* Bih  = dir ? Bbih : Bfih;
  const _Float16* Bhh  = dir ? Bbhh : Bfhh;
  const float*    bias = dir ? bbk : bf;
  const int colbase = dir ? HH : 0;

  const int tid = threadIdx.x, lane = tid & 31, wave = tid >> 5;
  for (int i = tid; i < BB * HPD; i += 1024) h16[i] = (_Float16)0.f;
  for (int i = tid; i < BB * H2P; i += 1024) z16[i] = (_Float16)0.f;
  float creg[5];
#pragma unroll
  for (int k = 0; k < 5; ++k) creg[k] = 0.f;
  __syncthreads();

  for (int s = 0; s < LLP; ++s) {
    const int t = dir ? (LLP - 1 - s) : s;
    // A: hWr = h @ Wr^T (2x10 tiles)
    for (int tile = wave; tile < 2 * HPT; tile += 32) {
      int mt = tile / HPT, nt = tile - mt * HPT;
      v8f acc = {};
#pragma unroll
      for (int kt = 0; kt < KT_H; ++kt)
        acc = wmma_f16(load_frag_pk(h16, KT_H, mt, kt, lane),
                       load_frag_pk(Bwr, KT_H, nt, kt, lane), acc);
      int col = (nt << 4) + (lane & 15);
      int rb  = (mt << 4) + ((lane >> 4) << 3);
#pragma unroll
      for (int r = 0; r < 8; ++r) hWr[(rb + r) * HPD + col] = acc[r];
    }
    __syncthreads();
    // B: scores s[l][b] = sum_j tanh(aq + HpWp_t + hWr)*wa + ba
    for (int p = tid; p < LLQ * BB; p += 1024) {
      int b = p & 31;
      float acc = ba[0];
      const float* aqrow = aq   + (size_t)p * HPD;
      const float* hprow = HpWp + ((size_t)t * BB + b) * HPD;
      const float* hwrow = hWr  + b * HPD;
      for (int j = 0; j < HH; ++j)
        acc += tanhf(aqrow[j] + hprow[j] + hwrow[j]) * wa[j];
      sco[p] = acc;
    }
    __syncthreads();
    // C: softmax over question positions
    if (tid < BB) {
      int b = tid;
      float mx = -3.0e38f;
      for (int l = 0; l < LLQ; ++l) mx = fmaxf(mx, sco[l * BB + b]);
      float sum = 0.f;
      for (int l = 0; l < LLQ; ++l) sum += __expf(sco[l * BB + b] - mx);
      float inv = 1.f / sum;
      for (int l = 0; l < LLQ; ++l)
        sco[l * BB + b] = __expf(sco[l * BB + b] - mx) * inv;
    }
    __syncthreads();
    // D: wq = sum_l alpha*Hq ; z = [hp, wq] * mask (packed layout)
    for (int p = tid; p < BB * HH; p += 1024) {
      int j = p >> 5, b = p & 31;
      float acc = 0.f;
      for (int l = 0; l < LLQ; ++l)
        acc += sco[l * BB + b] * Hq32[((size_t)l * BB + b) * HPD + j];
      float m  = (t < plens[b]) ? 1.f : 0.f;
      float hp = (float)Hp16[((size_t)t * BB + b) * HPD + j];
      z16[apack_idx(b, j, KT_2H)]      = (_Float16)(hp * m);
      z16[apack_idx(b, HH + j, KT_2H)] = (_Float16)(acc * m);
    }
    __syncthreads();
    // E: gates = z@Wih^T + h@Whh^T + b (76 tiles, 15 K-steps)
    for (int tile = wave; tile < 2 * G4T; tile += 32) {
      int mt = tile / G4T, nt = tile - mt * G4T;
      v8f acc = {};
#pragma unroll
      for (int kt = 0; kt < KT_2H; ++kt)
        acc = wmma_f16(load_frag_pk(z16, KT_2H, mt, kt, lane),
                       load_frag_pk(Bih, KT_2H, nt, kt, lane), acc);
#pragma unroll
      for (int kt = 0; kt < KT_H; ++kt)
        acc = wmma_f16(load_frag_pk(h16, KT_H, mt, kt, lane),
                       load_frag_pk(Bhh, KT_H, nt, kt, lane), acc);
      int col = (nt << 4) + (lane & 15);
      int rb  = (mt << 4) + ((lane >> 4) << 3);
      float bv = (col < 4 * HH) ? bias[col] : 0.f;
#pragma unroll
      for (int r = 0; r < 8; ++r) gates[(rb + r) * G4 + col] = acc[r] + bv;
    }
    __syncthreads();
    // F: masked LSTM pointwise
#pragma unroll
    for (int k = 0; k < 5; ++k) {
      int p = tid + (k << 10);
      if (p < BB * HH) {
        int j = p >> 5, b = p & 31;
        float gi = gates[b * G4 + j];
        float gf = gates[b * G4 + HH + j];
        float gg = gates[b * G4 + 2 * HH + j];
        float go = gates[b * G4 + 3 * HH + j];
        float c2 = sigmf(gf) * creg[k] + sigmf(gi) * tanhf(gg);
        float hv = sigmf(go) * tanhf(c2);
        float m  = (t < plens[b]) ? 1.f : 0.f;
        c2 *= m; hv *= m;
        creg[k] = c2;
        h16[apack_idx(b, j, KT_H)] = (_Float16)hv;
        Hr16[((size_t)t * BB + b) * H2P + colbase + j] = (_Float16)hv;
      }
    }
    __syncthreads();
  }
}

// ---------------------------------------------------------------------------
// Answer pointer: 2 iterations (start/end). 1 block, 1024 threads.
// Scores + softmax live in d_out.
// ---------------------------------------------------------------------------
__global__ void __launch_bounds__(1024)
k_answer(const float* __restrict__ am, const _Float16* __restrict__ Hr16,
         const _Float16* __restrict__ Bwa,
         const _Float16* __restrict__ Bapih, const _Float16* __restrict__ Baphh,
         const float* __restrict__ ba_ans, const float* __restrict__ wb,
         const float* __restrict__ bb, const float* __restrict__ apb,
         float* __restrict__ out) {
  extern __shared__ char smem[];
  float*    haWa  = (float*)smem;                              // 32*HPD
  float*    gates = (float*)(smem + BB * HPD * 4);             // 32*G4
  _Float16* h16   = (_Float16*)(smem + BB * HPD * 4 + BB * G4 * 4);
  _Float16* z16   = (_Float16*)((char*)h16 + BB * HPD * 2);    // 32*H2P

  const int tid = threadIdx.x, lane = tid & 31, wave = tid >> 5;
  for (int i = tid; i < BB * HPD; i += 1024) h16[i] = (_Float16)0.f;
  for (int i = tid; i < BB * H2P; i += 1024) z16[i] = (_Float16)0.f;
  float creg[5];
#pragma unroll
  for (int k = 0; k < 5; ++k) creg[k] = 0.f;
  __syncthreads();

  for (int it = 0; it < 2; ++it) {
    // A: haWa = ha @ Wa^T + ba_ans
    for (int tile = wave; tile < 2 * HPT; tile += 32) {
      int mt = tile / HPT, nt = tile - mt * HPT;
      v8f acc = {};
#pragma unroll
      for (int kt = 0; kt < KT_H; ++kt)
        acc = wmma_f16(load_frag_pk(h16, KT_H, mt, kt, lane),
                       load_frag_pk(Bwa, KT_H, nt, kt, lane), acc);
      int col = (nt << 4) + (lane & 15);
      int rb  = (mt << 4) + ((lane >> 4) << 3);
      float bv = (col < HH) ? ba_ans[col] : 0.f;
#pragma unroll
      for (int r = 0; r < 8; ++r) haWa[(rb + r) * HPD + col] = acc[r] + bv;
    }
    __syncthreads();
    // B: raw scores -> out[it]
    float* dist = out + it * LLP * BB;
    for (int p = tid; p < LLP * BB; p += 1024) {
      int b = p & 31;
      float acc = bb[0];
      const float* amrow = am + (size_t)p * HPD;
      const float* harow = haWa + b * HPD;
      for (int j = 0; j < HH; ++j)
        acc += tanhf(amrow[j] + harow[j]) * wb[j];
      dist[p] = acc;
    }
    __syncthreads();
    // C: softmax over passage positions, in place in d_out
    if (tid < BB) {
      int b = tid;
      float mx = -3.0e38f;
      for (int l = 0; l < LLP; ++l) mx = fmaxf(mx, dist[l * BB + b]);
      float sum = 0.f;
      for (int l = 0; l < LLP; ++l) sum += __expf(dist[l * BB + b] - mx);
      float inv = 1.f / sum;
      for (int l = 0; l < LLP; ++l)
        dist[l * BB + b] = __expf(dist[l * BB + b] - mx) * inv;
    }
    __syncthreads();
    // D: wHr = sum_l beta * Hr -> z (packed)
    for (int p = tid; p < BB * 2 * HH; p += 1024) {
      int j = p >> 5, b = p & 31;   // j in [0,300)
      float acc = 0.f;
      for (int l = 0; l < LLP; ++l)
        acc += dist[l * BB + b] * (float)Hr16[((size_t)l * BB + b) * H2P + j];
      z16[apack_idx(b, j, KT_2H)] = (_Float16)acc;
    }
    __syncthreads();
    // E: gates = z@ap_Wih^T + ha@ap_Whh^T + ap_b
    for (int tile = wave; tile < 2 * G4T; tile += 32) {
      int mt = tile / G4T, nt = tile - mt * G4T;
      v8f acc = {};
#pragma unroll
      for (int kt = 0; kt < KT_2H; ++kt)
        acc = wmma_f16(load_frag_pk(z16, KT_2H, mt, kt, lane),
                       load_frag_pk(Bapih, KT_2H, nt, kt, lane), acc);
#pragma unroll
      for (int kt = 0; kt < KT_H; ++kt)
        acc = wmma_f16(load_frag_pk(h16, KT_H, mt, kt, lane),
                       load_frag_pk(Baphh, KT_H, nt, kt, lane), acc);
      int col = (nt << 4) + (lane & 15);
      int rb  = (mt << 4) + ((lane >> 4) << 3);
      float bv = (col < 4 * HH) ? apb[col] : 0.f;
#pragma unroll
      for (int r = 0; r < 8; ++r) gates[(rb + r) * G4 + col] = acc[r] + bv;
    }
    __syncthreads();
    // F: LSTM pointwise (no mask)
#pragma unroll
    for (int k = 0; k < 5; ++k) {
      int p = tid + (k << 10);
      if (p < BB * HH) {
        int j = p >> 5, b = p & 31;
        float gi = gates[b * G4 + j];
        float gf = gates[b * G4 + HH + j];
        float gg = gates[b * G4 + 2 * HH + j];
        float go = gates[b * G4 + 3 * HH + j];
        float c2 = sigmf(gf) * creg[k] + sigmf(gi) * tanhf(gg);
        float hv = sigmf(go) * tanhf(c2);
        creg[k] = c2;
        h16[apack_idx(b, j, KT_H)] = (_Float16)hv;
      }
    }
    __syncthreads();
  }
}

// ---------------------------------------------------------------------------
// Host orchestration
// ---------------------------------------------------------------------------
extern "C" void kernel_launch(void* const* d_in, const int* in_sizes, int n_in,
                              void* d_out, int out_size, void* d_ws, size_t ws_size,
                              hipStream_t stream) {
  (void)in_sizes; (void)n_in; (void)out_size; (void)ws_size;

  const int*   pids      = (const int*)d_in[0];
  const int*   qids      = (const int*)d_in[1];
  const int*   plens     = (const int*)d_in[2];
  const int*   qlens     = (const int*)d_in[3];
  const float* emb       = (const float*)d_in[4];
  const float* plstm_Wih = (const float*)d_in[5];
  const float* plstm_Whh = (const float*)d_in[6];
  const float* plstm_b   = (const float*)d_in[7];
  const float* Wq        = (const float*)d_in[8];
  const float* Wp        = (const float*)d_in[9];
  const float* bp        = (const float*)d_in[10];
  const float* Wr        = (const float*)d_in[11];
  const float* wa        = (const float*)d_in[12];
  const float* ba        = (const float*)d_in[13];
  const float* mf_Wih    = (const float*)d_in[14];
  const float* mf_Whh    = (const float*)d_in[15];
  const float* mf_b      = (const float*)d_in[16];
  const float* mb_Wih    = (const float*)d_in[17];
  const float* mb_Whh    = (const float*)d_in[18];
  const float* mb_b      = (const float*)d_in[19];
  const float* Vm        = (const float*)d_in[20];
  const float* Wa_ans    = (const float*)d_in[21];
  const float* ba_ans    = (const float*)d_in[22];
  const float* wb        = (const float*)d_in[23];
  const float* bb        = (const float*)d_in[24];
  const float* ap_Wih    = (const float*)d_in[25];
  const float* ap_Whh    = (const float*)d_in[26];
  const float* ap_b      = (const float*)d_in[27];
  float* out = (float*)d_out;

  // ---- carve workspace ----
  char* ws = (char*)d_ws;
  size_t off = 0;
  auto a16 = [&](size_t elems) {
    _Float16* p = (_Float16*)(ws + off);
    off += (elems * 2 + 255) & ~(size_t)255;
    return p;
  };
  auto a32 = [&](size_t elems) {
    float* p = (float*)(ws + off);
    off += (elems * 4 + 255) & ~(size_t)255;
    return p;
  };

  const size_t NTP = (size_t)LLP * BB;   // 12800 passage tokens
  const size_t NTQ = (size_t)LLQ * BB;   // 1600 question tokens

  _Float16* embP16 = a16(NTP * EP);
  _Float16* embQ16 = a16(NTQ * EP);
  _Float16* Bp_ih  = a16((size_t)EP * G4);
  _Float16* Bp_hh  = a16((size_t)HPD * G4);
  _Float16* Bwq    = a16((size_t)HPD * HPD);
  _Float16* Bwp    = a16((size_t)HPD * HPD);
  _Float16* Bwr    = a16((size_t)HPD * HPD);
  _Float16* Bmf_ih = a16((size_t)H2P * G4);
  _Float16* Bmf_hh = a16((size_t)HPD * G4);
  _Float16* Bmb_ih = a16((size_t)H2P * G4);
  _Float16* Bmb_hh = a16((size_t)HPD * G4);
  _Float16* Bvm    = a16((size_t)H2P * HPD);
  _Float16* Bwa    = a16((size_t)HPD * HPD);
  _Float16* Bap_ih = a16((size_t)H2P * G4);
  _Float16* Bap_hh = a16((size_t)HPD * G4);
  float*    XpW    = a32(NTP * G4);
  float*    XqW    = a32(NTQ * G4);
  _Float16* Hp16   = a16(NTP * HPD);
  _Float16* Hq16   = a16(NTQ * HPD);
  float*    Hq32f  = a32(NTQ * HPD);
  float*    aqb    = a32(NTQ * HPD);
  float*    HpWpb  = a32(NTP * HPD);
  _Float16* Hr16   = a16(NTP * H2P);
  float*    amb    = a32(NTP * HPD);

  const dim3 B256(256);
  const dim3 B1024(1024);
  auto g1 = [](size_t n) { return dim3((unsigned)((n + 255) / 256)); };
  auto gemmGrid = [](int M, int Npad) {
    int tiles = (M / 16) * (Npad / 16);
    return dim3((unsigned)((tiles + 7) / 8));   // 8 waves / block
  };

  // ---- 1. weight conversions (f32 -> fragment-packed f16 W^T) ----
  k_cvt_pack<<<g1((size_t)EP * G4),  B256, 0, stream>>>(plstm_Wih, Bp_ih, 600, 300, G4, EP);
  k_cvt_pack<<<g1((size_t)HPD * G4), B256, 0, stream>>>(plstm_Whh, Bp_hh, 600, 150, G4, HPD);
  k_cvt_pack<<<g1((size_t)HPD * HPD), B256, 0, stream>>>(Wq, Bwq, 150, 150, HPD, HPD);
  k_cvt_pack<<<g1((size_t)HPD * HPD), B256, 0, stream>>>(Wp, Bwp, 150, 150, HPD, HPD);
  k_cvt_pack<<<g1((size_t)HPD * HPD), B256, 0, stream>>>(Wr, Bwr, 150, 150, HPD, HPD);
  k_cvt_pack<<<g1((size_t)H2P * G4), B256, 0, stream>>>(mf_Wih, Bmf_ih, 600, 300, G4, H2P);
  k_cvt_pack<<<g1((size_t)HPD * G4), B256, 0, stream>>>(mf_Whh, Bmf_hh, 600, 150, G4, HPD);
  k_cvt_pack<<<g1((size_t)H2P * G4), B256, 0, stream>>>(mb_Wih, Bmb_ih, 600, 300, G4, H2P);
  k_cvt_pack<<<g1((size_t)HPD * G4), B256, 0, stream>>>(mb_Whh, Bmb_hh, 600, 150, G4, HPD);
  k_cvt_pack<<<g1((size_t)H2P * HPD), B256, 0, stream>>>(Vm, Bvm, 150, 300, HPD, H2P);
  k_cvt_pack<<<g1((size_t)HPD * HPD), B256, 0, stream>>>(Wa_ans, Bwa, 150, 150, HPD, HPD);
  k_cvt_pack<<<g1((size_t)H2P * G4), B256, 0, stream>>>(ap_Wih, Bap_ih, 600, 300, G4, H2P);
  k_cvt_pack<<<g1((size_t)HPD * G4), B256, 0, stream>>>(ap_Whh, Bap_hh, 600, 150, G4, HPD);

  // ---- 2. embedding gather + zero padded state buffers ----
  k_gather<<<g1(NTP * EP), B256, 0, stream>>>(pids, emb, embP16, (int)NTP);
  k_gather<<<g1(NTQ * EP), B256, 0, stream>>>(qids, emb, embQ16, (int)NTQ);
  k_zero16<<<g1(NTP * HPD), B256, 0, stream>>>(Hp16, (int)(NTP * HPD));
  k_zero16<<<g1(NTQ * HPD), B256, 0, stream>>>(Hq16, (int)(NTQ * HPD));
  k_zero16<<<g1(NTP * H2P), B256, 0, stream>>>(Hr16, (int)(NTP * H2P));

  // ---- 3. time-parallel input projections X @ Wih^T ----
  k_gemm<<<gemmGrid((int)NTP, G4), B256, 0, stream>>>(
      embP16, Bp_ih, XpW, (int)NTP, G4, EP, nullptr, 0);
  k_gemm<<<gemmGrid((int)NTQ, G4), B256, 0, stream>>>(
      embQ16, Bp_ih, XqW, (int)NTQ, G4, EP, nullptr, 0);

  // ---- 4. pre-encoding LSTM scan (passage + question in parallel) ----
  const size_t SM_PRE = (size_t)BB * HPD * 2 + (size_t)BB * G4 * 4;  // 88 KB
  k_prelstm<<<2, B1024, SM_PRE, stream>>>(XpW, XqW, Bp_hh, plstm_b,
                                          plens, qlens, Hp16, Hq16, Hq32f);

  // ---- 5. step-invariant attention projections ----
  k_gemm<<<gemmGrid((int)NTQ, HPD), B256, 0, stream>>>(
      Hq16, Bwq, aqb, (int)NTQ, HPD, HPD, nullptr, 0);               // aq
  k_gemm<<<gemmGrid((int)NTP, HPD), B256, 0, stream>>>(
      Hp16, Bwp, HpWpb, (int)NTP, HPD, HPD, bp, HH);                 // Hp@Wp^T+bp

  // ---- 6. match-LSTM scans (fwd + bwd in parallel) ----
  const size_t SM_MATCH = (size_t)BB * HPD * 4 + (size_t)BB * G4 * 4 +
                          (size_t)LLQ * BB * 4 + (size_t)BB * HPD * 2 +
                          (size_t)BB * H2P * 2;                      // ~132 KB
  k_match<<<2, B1024, SM_MATCH, stream>>>(aqb, HpWpb, Hq32f, Hp16, Bwr,
                                          Bmf_ih, Bmf_hh, mf_b,
                                          Bmb_ih, Bmb_hh, mb_b,
                                          wa, ba, plens, Hr16);

  // ---- 7. am = Hr @ Vm^T ----
  k_gemm<<<gemmGrid((int)NTP, HPD), B256, 0, stream>>>(
      Hr16, Bvm, amb, (int)NTP, HPD, H2P, nullptr, 0);

  // ---- 8. answer pointer (writes [2, LP, B] into d_out) ----
  const size_t SM_ANS = (size_t)BB * HPD * 4 + (size_t)BB * G4 * 4 +
                        (size_t)BB * HPD * 2 + (size_t)BB * H2P * 2; // ~126 KB
  k_answer<<<1, B1024, SM_ANS, stream>>>(amb, Hr16, Bwa, Bap_ih, Bap_hh,
                                         ba_ans, wb, bb, ap_b, out);
}
